// MultiScaleDiffusionAttention_53051436040625
// MI455X (gfx1250) — matrix operand order repaired and checked by reading, not compiled, not run on hardware
//
#include <hip/hip_runtime.h>
#include <hip/hip_bf16.h>

typedef __attribute__((ext_vector_type(16))) __bf16 v16bf;
typedef __attribute__((ext_vector_type(8)))  __bf16 v8bf;
typedef __attribute__((ext_vector_type(8)))  float  v8f;

#define WMMA_BF16(a,b,c) __builtin_amdgcn_wmma_f32_16x16x32_bf16(false,(a),false,(b),(short)0,(c),false,false)

static constexpr int Bsz = 8, Nn = 1024, Cc = 320, Hh = 5, Dd = 64;
static constexpr int SLD = 1028;   // f32 LDS row stride for score slab (pad vs banks)
static constexpr int PLD = 1040;   // bf16 LDS row stride for probabilities (16B multiple)

__device__ __forceinline__ int reflect_n(int r) {
    return r < 0 ? -r : (r > Nn - 1 ? 2 * (Nn - 1) - r : r);
}

// 16 contiguous bf16 -> v16bf (B operand: one column, 16 K-values)
__device__ __forceinline__ v16bf load16(const __bf16* p) {
    v8bf lo = *(const v8bf*)p, hi = *(const v8bf*)(p + 8);
    v16bf r;
#pragma unroll
    for (int i = 0; i < 8; ++i) { r[i] = lo[i]; r[i + 8] = hi[i]; }
    return r;
}

// A operand: two 8-elem chunks (K {base, base+8}) from one row
__device__ __forceinline__ v16bf loadA2(const __bf16* p0, const __bf16* p1) {
    v8bf a = *(const v8bf*)p0, b = *(const v8bf*)p1;
    v16bf r;
#pragma unroll
    for (int i = 0; i < 8; ++i) { r[i] = a[i]; r[i + 8] = b[i]; }
    return r;
}

__device__ __forceinline__ v8f vzero8() { v8f z = {0.f,0.f,0.f,0.f,0.f,0.f,0.f,0.f}; return z; }

// ---------------- Kernel 0: f32 -> bf16 convert ----------------
__global__ void msda_cvt_bf16(const float* __restrict__ src, __bf16* __restrict__ dst, int n) {
    int i = blockIdx.x * blockDim.x + threadIdx.x;
    if (i < n) dst[i] = (__bf16)src[i];
}

// ---------------- Kernel 1: LayerNorm -> bf16 ----------------
__global__ void msda_layernorm(const float* __restrict__ x, const float* __restrict__ g,
                               const float* __restrict__ b, __bf16* __restrict__ xn) {
    int row = blockIdx.x, tid = threadIdx.x;
    __shared__ float rs[256], rq[256];
    const float* xr = x + (size_t)row * Cc;
    float s1 = 0.f, s2 = 0.f;
    for (int i = tid; i < Cc; i += 256) { float v = xr[i]; s1 += v; s2 += v * v; }
    rs[tid] = s1; rq[tid] = s2; __syncthreads();
    for (int off = 128; off > 0; off >>= 1) {
        if (tid < off) { rs[tid] += rs[tid + off]; rq[tid] += rq[tid + off]; }
        __syncthreads();
    }
    float mu = rs[0] * (1.f / Cc);
    float var = rq[0] * (1.f / Cc) - mu * mu;
    float inv = rsqrtf(var + 1e-5f);
    for (int i = tid; i < Cc; i += 256)
        xn[(size_t)row * Cc + i] = (__bf16)((xr[i] - mu) * inv * g[i] + b[i]);
}

// ---------------- Kernel 2: QKV GEMM (WMMA bf16) + scatter Q/K/Vt ----------------
// M=8192, N=960, K=320. One 16x64 tile per wave (4 accumulators, 4x A reuse).
// N-quad tq covers o in [tq*64, tq*64+64): wave-uniform s = tq/5, head = tq%5.
__global__ void msda_qkv_gemm(const __bf16* __restrict__ xn, const __bf16* __restrict__ w,
                              const float* __restrict__ bias,
                              __bf16* __restrict__ qb, __bf16* __restrict__ kb,
                              __bf16* __restrict__ vtb) {
    const int lane = threadIdx.x & 31, wave = threadIdx.x >> 5;
    const int lo = lane & 15, hi = lane >> 4;
    const int tile = blockIdx.x * 8 + wave;       // 512*15 = 7680 wave-tiles
    const int tq = tile % 15, tm = tile / 15;
    const int rowA = tm * 16 + lo;

    v8f acc[4];
#pragma unroll
    for (int dn = 0; dn < 4; ++dn) acc[dn] = vzero8();

#pragma unroll 2
    for (int kk = 0; kk < Cc; kk += 32) {
        const __bf16* ap = xn + (size_t)rowA * Cc + kk + hi * 16;
        v16bf a = loadA2(ap, ap + 8);
#pragma unroll
        for (int dn = 0; dn < 4; ++dn) {
            int colB = tq * 64 + dn * 16 + lo;
            v16bf bm = load16(w + (size_t)colB * Cc + kk + hi * 16);
            acc[dn] = WMMA_BF16(a, bm, acc[dn]);
        }
    }

    const int s = tq / 5, hh = tq % 5;            // wave-uniform: q/k/v select + head
#pragma unroll
    for (int dn = 0; dn < 4; ++dn) {
        int dd = dn * 16 + lo;
        float bv = bias[tq * 64 + dd];
#pragma unroll
        for (int v = 0; v < 8; ++v) {
            int tok = tm * 16 + v + hi * 8;
            float val = acc[dn][v] + bv;
            int b0 = tok >> 10, n = tok & 1023;
            size_t bh = (size_t)(b0 * Hh + hh);
            if (s == 0)      qb[(bh * Nn + n) * Dd + dd] = (__bf16)(val * 0.125f); // fold d^-0.5
            else if (s == 1) kb[(bh * Nn + n) * Dd + dd] = (__bf16)val;
            else             vtb[(bh * Dd + dd) * Nn + n] = (__bf16)val;           // V transposed
        }
    }
}

// ---------------- Kernel 3: fused attention (QK^T + blur + softmax + PV) ----------------
// grid (64, 40): 16-row query tile per block, bh per blockIdx.y. 256 threads = 8 waves.
__global__ void msda_attn(const __bf16* __restrict__ qb, const __bf16* __restrict__ kb,
                          const __bf16* __restrict__ vtb, __bf16* __restrict__ aout) {
    extern __shared__ char smem[];
    float* S  = (float*)smem;                 // 22 x SLD : score halo slab (f32)
    float* Vb = S + 22 * SLD;                 // 16 x SLD : vertical blur pass
    float* O  = Vb + 16 * SLD;                // 16 x 68  : PV reduction
    __bf16* P = (__bf16*)(O + 16 * 68);       // 16 x PLD : probabilities (bf16)

    const int tid = threadIdx.x;
    const int lane = tid & 31, wave = tid >> 5;
    const int lo = lane & 15, hi = lane >> 4;
    const int bh = blockIdx.y;
    const int h = bh % Hh;
    const int b0 = bh / Hh;
    const int n0 = blockIdx.x * 16;

    // normalized separable Gaussian taps for this head
    const float sig = (float)(h + 1);
    const float alpha = 0.1f * (float)(h + 1);
    float gn[7]; float sg = 0.f;
#pragma unroll
    for (int i = 0; i < 7; ++i) { float xx = (float)(i - 3); gn[i] = __expf(-xx * xx / (2.f * sig * sig)); sg += gn[i]; }
#pragma unroll
    for (int i = 0; i < 7; ++i) gn[i] /= sg;

    for (int i = tid; i < 16 * 68; i += 256) O[i] = 0.f;   // zero PV accumulator

    const __bf16* qbase = qb + (size_t)bh * Nn * Dd;
    const __bf16* kbase = kb + (size_t)bh * Nn * Dd;
    const __bf16* vbase = vtb + (size_t)bh * Dd * Nn;

    // warm caches for stage 4's Vt slab (64x1024 bf16 = 128KB) while QK^T runs
#pragma unroll
    for (int pf = 0; pf < 4; ++pf)
        __builtin_prefetch((const char*)vbase + (size_t)(tid + pf * 256) * 128, 0, 1);

    // ---- Stage 1: S[j][m] = (Q K^T)[reflect(n0-3+j)][m], j = 0..21 ----
#pragma unroll
    for (int rt = 0; rt < 2; ++rt) {
        v8f acc[8];
#pragma unroll
        for (int ct = 0; ct < 8; ++ct) acc[ct] = vzero8();
#pragma unroll
        for (int kk = 0; kk < Dd; kk += 32) {
            int j = rt * 16 + lo; if (j > 21) j = 21;        // pad rows (discarded)
            int r = reflect_n(n0 - 3 + j);                   // gathered A rows
            const __bf16* qp = qbase + (size_t)r * Dd + kk + hi * 16;
            v16bf a = loadA2(qp, qp + 8);
#pragma unroll
            for (int ct = 0; ct < 8; ++ct) {
                int m = wave * 128 + ct * 16 + lo;
                v16bf bm = load16(kbase + (size_t)m * Dd + kk + hi * 16);
                acc[ct] = WMMA_BF16(a, bm, acc[ct]);
            }
        }
#pragma unroll
        for (int ct = 0; ct < 8; ++ct)
#pragma unroll
            for (int v = 0; v < 8; ++v) {
                int j = rt * 16 + v + hi * 8;
                if (j < 22) S[j * SLD + wave * 128 + ct * 16 + lo] = acc[ct][v];
            }
    }
    __syncthreads();

    // ---- Stage 2: vertical 7-tap over the 22-row halo ----
    const int row = tid >> 4;            // 0..15 (half-wave per row)
    const int cb = (tid & 15) * 64;      // 64 columns per thread
    for (int m = 0; m < 64; ++m) {
        int mm = cb + m;
        float acc = 0.f;
#pragma unroll
        for (int dy = 0; dy < 7; ++dy) acc += gn[dy] * S[(row + dy) * SLD + mm];
        Vb[row * SLD + mm] = acc;
    }
    __syncthreads();

    // ---- Stage 3: horizontal 7-tap (reflect cols), logits, softmax ----
    float mx = -1e30f;
    for (int m = 0; m < 64; ++m) {
        int mm = cb + m;
        float bl = 0.f;
#pragma unroll
        for (int dx = 0; dx < 7; ++dx) {
            int c = mm + dx - 3;
            c = c < 0 ? -c : (c > Nn - 1 ? 2 * (Nn - 1) - c : c);
            bl += gn[dx] * Vb[row * SLD + c];
        }
        float lg = S[(row + 3) * SLD + mm] + alpha * bl;
        S[(row + 3) * SLD + mm] = lg;                 // only this thread touches it
        mx = fmaxf(mx, lg);
    }
    mx = fmaxf(mx, __shfl_xor(mx, 1));
    mx = fmaxf(mx, __shfl_xor(mx, 2));
    mx = fmaxf(mx, __shfl_xor(mx, 4));
    mx = fmaxf(mx, __shfl_xor(mx, 8));                // half-wave = one row
    float sum = 0.f;
    for (int m = 0; m < 64; ++m) {
        int mm = cb + m;
        float e = __expf(S[(row + 3) * SLD + mm] - mx);
        sum += e;
        P[row * PLD + mm] = (__bf16)e;
    }
    sum += __shfl_xor(sum, 1);
    sum += __shfl_xor(sum, 2);
    sum += __shfl_xor(sum, 4);
    sum += __shfl_xor(sum, 8);
    float sc = 1.f / sum;
    for (int m = 0; m < 64; ++m) {
        int mm = cb + m;
        P[row * PLD + mm] = (__bf16)((float)P[row * PLD + mm] * sc);
    }
    __syncthreads();

    // ---- Stage 4: PV. Each wave owns 128 m-values; reduce via LDS atomics ----
    v8f acc2[4];
#pragma unroll
    for (int dt = 0; dt < 4; ++dt) acc2[dt] = vzero8();
#pragma unroll
    for (int c = 0; c < 4; ++c) {
        int mbase = wave * 128 + c * 32;
        const __bf16* pp = P + (size_t)lo * PLD + mbase + hi * 16;   // A from LDS
        v16bf a = loadA2(pp, pp + 8);
#pragma unroll
        for (int dt = 0; dt < 4; ++dt) {
            int dd = dt * 16 + lo;
            v16bf bm = load16(vbase + (size_t)dd * Nn + mbase + hi * 16);
            acc2[dt] = WMMA_BF16(a, bm, acc2[dt]);
        }
    }
#pragma unroll
    for (int dt = 0; dt < 4; ++dt)
#pragma unroll
        for (int v = 0; v < 8; ++v)
            atomicAdd(&O[(v + hi * 8) * 68 + dt * 16 + lo], acc2[dt][v]);
    __syncthreads();

    // ---- write (B,N,C) head slice as bf16 ----
    for (int i = tid; i < 16 * 64; i += 256) {
        int rw = i >> 6, dd = i & 63;
        size_t tok = (size_t)b0 * Nn + n0 + rw;
        aout[tok * Cc + h * 64 + dd] = (__bf16)O[rw * 68 + dd];
    }
}

// ---------------- Kernel 4: output projection (WMMA bf16) -> f32 ----------------
// M=8192, N=320, K=320. One 16x64 tile per wave (4 accumulators).
__global__ void msda_proj_gemm(const __bf16* __restrict__ ao, const __bf16* __restrict__ w,
                               const float* __restrict__ bias, float* __restrict__ out) {
    const int lane = threadIdx.x & 31, wave = threadIdx.x >> 5;
    const int lo = lane & 15, hi = lane >> 4;
    const int tile = blockIdx.x * 8 + wave;       // 512*5 = 2560 wave-tiles
    const int tq = tile % 5, tm = tile / 5;
    const int rowA = tm * 16 + lo;

    v8f acc[4];
#pragma unroll
    for (int dn = 0; dn < 4; ++dn) acc[dn] = vzero8();

#pragma unroll 2
    for (int kk = 0; kk < Cc; kk += 32) {
        const __bf16* ap = ao + (size_t)rowA * Cc + kk + hi * 16;
        v16bf a = loadA2(ap, ap + 8);
#pragma unroll
        for (int dn = 0; dn < 4; ++dn) {
            int colB = tq * 64 + dn * 16 + lo;
            v16bf bm = load16(w + (size_t)colB * Cc + kk + hi * 16);
            acc[dn] = WMMA_BF16(a, bm, acc[dn]);
        }
    }
#pragma unroll
    for (int dn = 0; dn < 4; ++dn) {
        int o = tq * 64 + dn * 16 + lo;
        float bv = bias[o];
#pragma unroll
        for (int v = 0; v < 8; ++v) {
            int tok = tm * 16 + v + hi * 8;
            out[(size_t)tok * Cc + o] = acc[dn][v] + bv;
        }
    }
}

extern "C" void kernel_launch(void* const* d_in, const int* in_sizes, int n_in,
                              void* d_out, int out_size, void* d_ws, size_t ws_size,
                              hipStream_t stream) {
    const float* x      = (const float*)d_in[0];
    const float* ln_g   = (const float*)d_in[1];
    const float* ln_b   = (const float*)d_in[2];
    const float* qkv_w  = (const float*)d_in[3];
    const float* qkv_b  = (const float*)d_in[4];
    const float* proj_w = (const float*)d_in[5];
    const float* proj_b = (const float*)d_in[6];
    float* out = (float*)d_out;

    // workspace layout (bf16 buffers, 256B aligned)
    char* ws = (char*)d_ws;
    size_t off = 0;
    auto carve = [&](size_t elems) {
        __bf16* p = (__bf16*)(ws + off);
        off += (elems * sizeof(__bf16) + 255) & ~(size_t)255;
        return p;
    };
    const size_t TOK = (size_t)Bsz * Nn;                       // 8192
    const size_t QKV_ELEMS = (size_t)Bsz * Hh * Nn * Dd;       // 2,621,440
    __bf16* xn_bf    = carve(TOK * Cc);
    __bf16* qkvw_bf  = carve((size_t)3 * Cc * Cc);
    __bf16* projw_bf = carve((size_t)Cc * Cc);
    __bf16* qbuf     = carve(QKV_ELEMS);
    __bf16* kbuf     = carve(QKV_ELEMS);
    __bf16* vtbuf    = carve(QKV_ELEMS);
    __bf16* aout     = carve(TOK * Cc);
    (void)ws_size; (void)in_sizes; (void)n_in; (void)out_size;

    // 0) convert weights to bf16
    msda_cvt_bf16<<<(3 * Cc * Cc + 255) / 256, 256, 0, stream>>>(qkv_w, qkvw_bf, 3 * Cc * Cc);
    msda_cvt_bf16<<<(Cc * Cc + 255) / 256, 256, 0, stream>>>(proj_w, projw_bf, Cc * Cc);

    // 1) LayerNorm
    msda_layernorm<<<(int)TOK, 256, 0, stream>>>(x, ln_g, ln_b, xn_bf);

    // 2) QKV GEMM + scatter: 512 M-tiles * 15 N-quads = 7680 wave-tiles / 8 waves
    msda_qkv_gemm<<<960, 256, 0, stream>>>(xn_bf, qkvw_bf, qkv_b, qbuf, kbuf, vtbuf);

    // 3) fused attention: grid (64 row-tiles, 40 bh), ~190KB dynamic LDS
    size_t lds = (size_t)(22 * SLD + 16 * SLD + 16 * 68) * 4 + (size_t)16 * PLD * 2;
    msda_attn<<<dim3(64, Bsz * Hh), 256, lds, stream>>>(qbuf, kbuf, vtbuf, aout);

    // 4) output projection: 512 M-tiles * 5 N-quads = 2560 wave-tiles / 8 waves
    msda_proj_gemm<<<320, 256, 0, stream>>>(aout, projw_bf, proj_b, out);
}
